// TemporalRNN_15745350107207
// MI455X (gfx1250) — matrix operand order repaired and checked by reading, not compiled, not run on hardware
//
#include <hip/hip_runtime.h>
#include <hip/hip_bf16.h>
#include <math.h>

typedef __attribute__((ext_vector_type(16))) _Float16 v16h;
typedef __attribute__((ext_vector_type(8)))  _Float16 v8h;
typedef __attribute__((ext_vector_type(8)))  float    v8f;

#define B_SZ   256
#define T_SZ   1024
#define H_SZ   256
#define G_SZ   768          // 3*H
#define BT     16           // batch rows per workgroup (GRU)
#define HS_STR 264          // padded LDS row stride (halves), multiple of 8

__device__ __forceinline__ v8f wmma_f16(v16h a, v16h b, v8f c) {
    return __builtin_amdgcn_wmma_f32_16x16x32_f16(false, a, false, b, (short)0, c, false, false);
}

__device__ __forceinline__ v8f splat8(float x) {
    v8f v;
#pragma unroll
    for (int i = 0; i < 8; ++i) v[i] = x;
    return v;
}

// Low 32 bits of a flat pointer into __shared__ == wave-relative LDS byte
// address (flat->LDS aperture truncation, ISA 10.2).
__device__ __forceinline__ unsigned lds_addr_of(const void* p) {
    return (unsigned)(size_t)p;
}

__device__ __forceinline__ void async_ld_b128(const void* gsrc, void* ldst) {
    unsigned d = lds_addr_of(ldst);
    unsigned long long s = (unsigned long long)(size_t)gsrc;
    asm volatile("global_load_async_to_lds_b128 %0, %1, off"
                 :: "v"(d), "v"(s) : "memory");
}
__device__ __forceinline__ void async_ld_b128_off16(const void* gsrc, void* ldst) {
    unsigned d = lds_addr_of(ldst);
    unsigned long long s = (unsigned long long)(size_t)gsrc;
    asm volatile("global_load_async_to_lds_b128 %0, %1, off offset:16"
                 :: "v"(d), "v"(s) : "memory");
}
__device__ __forceinline__ void async_ld_b32(const void* gsrc, void* ldst) {
    unsigned d = lds_addr_of(ldst);
    unsigned long long s = (unsigned long long)(size_t)gsrc;
    asm volatile("global_load_async_to_lds_b32 %0, %1, off"
                 :: "v"(d), "v"(s) : "memory");
}
__device__ __forceinline__ void wait_asynccnt0() {
    asm volatile("s_wait_asynccnt 0x0" ::: "memory");
}

// A fragment (16x32 f16, M=16 rows, K=32): lane row M=l16; lanes 0-15 hold
// K = kb..kb+15; lanes 16-31 hold K = kb+16..kb+31.
__device__ __forceinline__ v16h load_a_lds(const _Float16* base, int stride,
                                           int l16, int kb, int hi) {
    union { v16h v; v8h h[2]; } u;
    const _Float16* p = base + l16 * stride + kb + hi * 16;
    u.h[0] = *(const v8h*)(p);
    u.h[1] = *(const v8h*)(p + 8);
    return u.v;
}

// B fragment (32x16 f16): lane column N=l16; lanes 0-15 K=kb..kb+15,
// lanes 16-31 K=kb+16..kb+31 (16 contiguous halves -> 32B aligned load).
__device__ __forceinline__ v16h load_b_glb(const _Float16* w, int kdim,
                                           int col, int kb, int hi) {
    return *(const v16h*)(w + (size_t)col * kdim + kb + hi * 16);
}

__device__ __forceinline__ float sigmoid_f(float x) {
    return 1.0f / (1.0f + __expf(-x));
}
__device__ __forceinline__ float tanh_f(float x) {
    float e = __expf(-2.0f * fabsf(x));
    float t = (1.0f - e) / (1.0f + e);
    return copysignf(t, x);
}

__global__ void cvt_f32_to_f16(const float* __restrict__ s,
                               _Float16* __restrict__ d, int n) {
    int i = blockIdx.x * 256 + threadIdx.x;
    if (i < n) d[i] = (_Float16)s[i];
}

// ---------------------------------------------------------------------------
// GRU layer kernel. grid = B/16 workgroups, block = 256 threads (8 waves).
// Each workgroup owns 16 batch rows for the whole time scan; each wave owns
// 32 hidden units (6 WMMA accumulators r/z/nh + 2 for x-side n).
// Input-side staging is double-buffered through LDS with async-to-LDS loads
// (ASYNCcnt), overlapping the t+1 stage with the t GEMM.
// ---------------------------------------------------------------------------
template <int LAYER>
__global__ void __launch_bounds__(256)
gru_layer_kernel(const float* __restrict__ x,        // [B,T,6]   (LAYER 0)
                 const _Float16* __restrict__ h_in,  // [B,T,256] (LAYER 1)
                 const _Float16* __restrict__ Whh,   // [768,256] f16
                 const _Float16* __restrict__ Wih1,  // [768,256] f16 (LAYER 1)
                 const float* __restrict__ Wih0,     // [768,6]   (LAYER 0)
                 const float* __restrict__ b_ih,     // [768]
                 const float* __restrict__ b_hh,     // [768]
                 _Float16* __restrict__ h_out,       // [B,T,256] f16 (LAYER 0)
                 float* __restrict__ out_f32)        // [B,T,256] f32 (LAYER 1)
{
    __shared__ _Float16 h_sh[BT * HS_STR];           // hidden state tile (f16)
    __shared__ _Float16 g_sh2[2][BT * HS_STR];       // staged input tiles (L1)
    __shared__ float    x_sh2[2][BT * 8];            // staged x_t tiles   (L0)

    const int tid  = threadIdx.x;
    const int wave = tid >> 5;
    const int lane = tid & 31;
    const int l16  = lane & 15;
    const int hi   = lane >> 4;
    const int b0   = blockIdx.x * BT;

    for (int i = tid; i < BT * HS_STR; i += 256) h_sh[i] = (_Float16)0.0f;

    float hreg[2][8];
#pragma unroll
    for (int s = 0; s < 2; ++s)
#pragma unroll
        for (int v = 0; v < 8; ++v) hreg[s][v] = 0.0f;

    int jj[2];
    jj[0] = wave * 32 + l16;
    jj[1] = wave * 32 + 16 + l16;

    float brz[2][2], bnh[2], bnx[2];
#pragma unroll
    for (int s = 0; s < 2; ++s) {
        brz[0][s] = b_ih[jj[s]]       + b_hh[jj[s]];
        brz[1][s] = b_ih[256 + jj[s]] + b_hh[256 + jj[s]];
        bnh[s]    = b_hh[512 + jj[s]];
        bnx[s]    = b_ih[512 + jj[s]];
    }

    float wih[3][2][6];
    if constexpr (LAYER == 0) {
#pragma unroll
        for (int g = 0; g < 3; ++g)
#pragma unroll
            for (int s = 0; s < 2; ++s)
#pragma unroll
                for (int i = 0; i < 6; ++i)
                    wih[g][s][i] = Wih0[(g * 256 + jj[s]) * 6 + i];
    }

    // async stage of timestep t into LDS buffer `buf`
    auto stage_issue = [&](int t, int buf) {
        if constexpr (LAYER == 0) {
            if (tid < BT * 6) {
                int row = tid / 6, i = tid % 6;
                const float* src = x + ((size_t)(b0 + row) * T_SZ + t) * 6 + i;
                async_ld_b32(src, &x_sh2[buf][row * 8 + i]);
            }
        } else {
            int row = tid >> 4, chunk = tid & 15;
            const _Float16* src =
                h_in + ((size_t)(b0 + row) * T_SZ + t) * H_SZ + chunk * 16;
            _Float16* dst = &g_sh2[buf][row * HS_STR + chunk * 16];
            async_ld_b128(src, dst);          // bytes 0..15 (offset applies to
            async_ld_b128_off16(src, dst);    // both sides) bytes 16..31
        }
    };

    stage_issue(0, 0);
    __syncthreads();          // h_sh zero-init visible

    for (int t = 0; t < T_SZ; ++t) {
        const int buf = t & 1;
        wait_asynccnt0();     // tile t resident in LDS (per-wave ASYNCcnt)
        __syncthreads();      // B1: all waves' stages + h_sh(t-1) visible

        // prefetch tile t+1 into the other buffer (overlaps the GEMM below;
        // that buffer's last readers finished before B2 of step t-1)
        if (t + 1 < T_SZ) stage_issue(t + 1, buf ^ 1);

        // ---- gate GEMMs: accA[g*2+s] (r,z,nh), accNX[s] (x-side n) -------
        v8f accA[6], accNX[2];
#pragma unroll
        for (int s = 0; s < 2; ++s) {
            accA[s]     = splat8(brz[0][s]);
            accA[2 + s] = splat8(brz[1][s]);
            accA[4 + s] = splat8(bnh[s]);
            accNX[s]    = splat8(bnx[s]);
        }

#pragma unroll
        for (int kc = 0; kc < 8; ++kc) {
            const int kb = kc * 32;
            v16h ah = load_a_lds(h_sh, HS_STR, l16, kb, hi);
            v16h ax;
            if constexpr (LAYER == 1)
                ax = load_a_lds(g_sh2[buf], HS_STR, l16, kb, hi);

            // load ALL B fragments of this K-chunk first (distinct regs ->
            // many loads in flight, WMMAs drain behind staggered waits)
            v16h bh[6];
#pragma unroll
            for (int g = 0; g < 3; ++g)
#pragma unroll
                for (int s = 0; s < 2; ++s)
                    bh[g * 2 + s] = load_b_glb(Whh, H_SZ, g * 256 + jj[s], kb, hi);

            v16h bx[6];
            if constexpr (LAYER == 1) {
#pragma unroll
                for (int g = 0; g < 3; ++g)
#pragma unroll
                    for (int s = 0; s < 2; ++s)
                        bx[g * 2 + s] =
                            load_b_glb(Wih1, H_SZ, g * 256 + jj[s], kb, hi);
            }

#pragma unroll
            for (int q = 0; q < 6; ++q)
                accA[q] = wmma_f16(ah, bh[q], accA[q]);
            if constexpr (LAYER == 1) {
#pragma unroll
                for (int q = 0; q < 4; ++q)
                    accA[q] = wmma_f16(ax, bx[q], accA[q]);
#pragma unroll
                for (int s = 0; s < 2; ++s)
                    accNX[s] = wmma_f16(ax, bx[4 + s], accNX[s]);
            }
        }

        // ---- layer-0 input side: 6-wide scalar dot ------------------------
        if constexpr (LAYER == 0) {
#pragma unroll
            for (int g = 0; g < 3; ++g)
#pragma unroll
                for (int s = 0; s < 2; ++s)
#pragma unroll
                    for (int v = 0; v < 8; ++v) {
                        const int m = v + hi * 8;
                        float d = 0.0f;
#pragma unroll
                        for (int i = 0; i < 6; ++i)
                            d = fmaf(x_sh2[buf][m * 8 + i], wih[g][s][i], d);
                        if (g < 2) accA[g * 2 + s][v] += d;
                        else       accNX[s][v] += d;
                    }
        }

        __syncthreads();      // B2: everyone done reading h_sh / input tile

        // ---- gate combine + state update ---------------------------------
#pragma unroll
        for (int s = 0; s < 2; ++s) {
            const int j = jj[s];
#pragma unroll
            for (int v = 0; v < 8; ++v) {
                const int m = v + hi * 8;
                float r = sigmoid_f(accA[s][v]);
                float z = sigmoid_f(accA[2 + s][v]);
                float n = tanh_f(accNX[s][v] + r * accA[4 + s][v]);
                float hn = (1.0f - z) * n + z * hreg[s][v];
                hreg[s][v] = hn;
                h_sh[m * HS_STR + j] = (_Float16)hn;
                size_t gi = ((size_t)(b0 + m) * T_SZ + t) * H_SZ + j;
                if constexpr (LAYER == 0) h_out[gi] = (_Float16)hn;
                else                      out_f32[gi] = hn;
            }
        }
        // next iteration's B1 orders these h_sh writes against K-loop reads
    }
}

// ---------------------------------------------------------------------------
// Readout: y = relu(out @ W1^T + b1) @ W2^T + b2, rows = B*T.
// block = 128 threads (4 waves); each wave owns an independent 16-row tile.
// ---------------------------------------------------------------------------
#define RO_WAVES 4
#define A_STR 264          // 256 + 8 pad
#define T_STR 136          // 128 + 8 pad
__global__ void __launch_bounds__(128)
readout_kernel(const float* __restrict__ outp,   // [B*T,256] f32
               const _Float16* __restrict__ W1h, // [128,256] f16
               const _Float16* __restrict__ W2h, // [80,128]  f16
               const float* __restrict__ b1,     // [128]
               const float* __restrict__ b2,     // [80]
               float* __restrict__ y)            // [B*T,80]
{
    __shared__ _Float16 a_sh[RO_WAVES * 16 * A_STR];
    __shared__ _Float16 t_sh[RO_WAVES * 16 * T_STR];

    const int tid  = threadIdx.x;
    const int wave = tid >> 5;
    const int lane = tid & 31;
    const int l16  = lane & 15;
    const int hi   = lane >> 4;
    const size_t row0 = ((size_t)blockIdx.x * RO_WAVES + wave) * 16;

    _Float16* aw = a_sh + wave * 16 * A_STR;
    _Float16* tw = t_sh + wave * 16 * T_STR;

    // stage 16x256 f32 rows -> f16 LDS
#pragma unroll 4
    for (int rr = 0; rr < 16; ++rr) {
        const float* src = outp + (row0 + rr) * H_SZ + lane * 8;
        float4 f0 = *(const float4*)(src);
        float4 f1 = *(const float4*)(src + 4);
        v8h h;
        h[0] = (_Float16)f0.x; h[1] = (_Float16)f0.y;
        h[2] = (_Float16)f0.z; h[3] = (_Float16)f0.w;
        h[4] = (_Float16)f1.x; h[5] = (_Float16)f1.y;
        h[6] = (_Float16)f1.z; h[7] = (_Float16)f1.w;
        *(v8h*)(aw + rr * A_STR + lane * 8) = h;
    }
    __syncthreads();

    // GEMM1: [16x256] x [256x128] -> relu -> t_sh (f16)
    v8f acc1[8];
#pragma unroll
    for (int n = 0; n < 8; ++n) acc1[n] = splat8(b1[n * 16 + l16]);
#pragma unroll
    for (int kc = 0; kc < 8; ++kc) {
        const int kb = kc * 32;
        v16h a = load_a_lds(aw, A_STR, l16, kb, hi);
        v16h bfr[8];
#pragma unroll
        for (int n = 0; n < 8; ++n)
            bfr[n] = load_b_glb(W1h, 256, n * 16 + l16, kb, hi);
#pragma unroll
        for (int n = 0; n < 8; ++n)
            acc1[n] = wmma_f16(a, bfr[n], acc1[n]);
    }
#pragma unroll
    for (int n = 0; n < 8; ++n)
#pragma unroll
        for (int v = 0; v < 8; ++v) {
            const int m = v + hi * 8;
            float r = fmaxf(acc1[n][v], 0.0f);
            tw[m * T_STR + n * 16 + l16] = (_Float16)r;
        }
    __syncthreads();

    // GEMM2: [16x128] x [128x80]
    v8f acc2[5];
#pragma unroll
    for (int n = 0; n < 5; ++n) acc2[n] = splat8(b2[n * 16 + l16]);
#pragma unroll
    for (int kc = 0; kc < 4; ++kc) {
        const int kb = kc * 32;
        v16h a = load_a_lds(tw, T_STR, l16, kb, hi);
        v16h bfr[5];
#pragma unroll
        for (int n = 0; n < 5; ++n)
            bfr[n] = load_b_glb(W2h, 128, n * 16 + l16, kb, hi);
#pragma unroll
        for (int n = 0; n < 5; ++n)
            acc2[n] = wmma_f16(a, bfr[n], acc2[n]);
    }
#pragma unroll
    for (int n = 0; n < 5; ++n)
#pragma unroll
        for (int v = 0; v < 8; ++v) {
            const int m = v + hi * 8;
            y[(row0 + m) * 80 + n * 16 + l16] = acc2[n][v];
        }
}

extern "C" void kernel_launch(void* const* d_in, const int* in_sizes, int n_in,
                              void* d_out, int out_size, void* d_ws, size_t ws_size,
                              hipStream_t stream) {
    (void)in_sizes; (void)n_in; (void)out_size; (void)ws_size;

    const float* x     = (const float*)d_in[0];
    const float* Wih0  = (const float*)d_in[1];
    const float* bih0  = (const float*)d_in[2];
    const float* Whh0  = (const float*)d_in[3];
    const float* bhh0  = (const float*)d_in[4];
    const float* Wih1  = (const float*)d_in[5];
    const float* bih1  = (const float*)d_in[6];
    const float* Whh1  = (const float*)d_in[7];
    const float* bhh1  = (const float*)d_in[8];
    const float* W1    = (const float*)d_in[9];
    const float* b1    = (const float*)d_in[10];
    const float* W2    = (const float*)d_in[11];
    const float* b2    = (const float*)d_in[12];

    float* y_out   = (float*)d_out;                                  // [B,T,80]
    float* out_f32 = (float*)d_out + (size_t)B_SZ * T_SZ * 80;       // [B,T,256]

    // workspace layout (halves)
    _Float16* h0_h   = (_Float16*)d_ws;                 // B*T*H halves
    _Float16* whh0_h = h0_h   + (size_t)B_SZ * T_SZ * H_SZ;
    _Float16* whh1_h = whh0_h + (size_t)G_SZ * H_SZ;
    _Float16* wih1_h = whh1_h + (size_t)G_SZ * H_SZ;
    _Float16* w1_h   = wih1_h + (size_t)G_SZ * H_SZ;
    _Float16* w2_h   = w1_h   + (size_t)128 * 256;

    cvt_f32_to_f16<<<(G_SZ * H_SZ + 255) / 256, 256, 0, stream>>>(Whh0, whh0_h, G_SZ * H_SZ);
    cvt_f32_to_f16<<<(G_SZ * H_SZ + 255) / 256, 256, 0, stream>>>(Whh1, whh1_h, G_SZ * H_SZ);
    cvt_f32_to_f16<<<(G_SZ * H_SZ + 255) / 256, 256, 0, stream>>>(Wih1, wih1_h, G_SZ * H_SZ);
    cvt_f32_to_f16<<<(128 * 256 + 255) / 256, 256, 0, stream>>>(W1, w1_h, 128 * 256);
    cvt_f32_to_f16<<<(80 * 128 + 255) / 256, 256, 0, stream>>>(W2, w2_h, 80 * 128);

    gru_layer_kernel<0><<<B_SZ / BT, 256, 0, stream>>>(
        x, nullptr, whh0_h, nullptr, Wih0, bih0, bhh0, h0_h, nullptr);

    gru_layer_kernel<1><<<B_SZ / BT, 256, 0, stream>>>(
        nullptr, h0_h, whh1_h, wih1_h, nullptr, bih1, bhh1, nullptr, out_f32);

    const size_t rows = (size_t)B_SZ * T_SZ;
    readout_kernel<<<(unsigned)(rows / (RO_WAVES * 16)), 128, 0, stream>>>(
        out_f32, w1_h, w2_h, b1, b2, y_out);
}
